// LowFrequency_77524159693275
// MI455X (gfx1250) — matrix operand order repaired and checked by reading, not compiled, not run on hardware
//
#include <hip/hip_runtime.h>
#include <hip/hip_bf16.h>

// ---------------------------------------------------------------------------
// MoE MLP, B=131072 rows, 3 experts, fused bf16-WMMA pipeline for gfx1250.
//   out  = blend4(relu(blend3(relu(blend2(relu(blend1(x)))))))
//   blend_L(h)[b,n] = sum_e gate[b,e] * (W_Le h[b] + b_Le)[n]
// ---------------------------------------------------------------------------

typedef unsigned short u16;
typedef unsigned int   u32;
typedef __attribute__((ext_vector_type(16))) __bf16 v16bf;
typedef __attribute__((ext_vector_type(8)))  float  v8f;

#define B_TOT 131072
#define MT    32          // rows per block in the fused kernel
#define LDA1  648         // padded LDS stride for K=640 (648*2B = 324 dwords, %64 = 4 -> conflict-free)
#define LDA2  520         // padded LDS stride for K=512

__device__ __forceinline__ u16 f2bf_rne(float f) {
  u32 u = __float_as_uint(f);
  u32 r = u + 0x7FFFu + ((u >> 16) & 1u);   // round-to-nearest-even
  return (u16)(r >> 16);
}

union ABits { uint4 q[2]; v16bf v; };

// A fragment (16x32 bf16, rows striped over lanes): lane L holds row L%16,
// K-chunks [k0 + (L>=16?8:0) .. +7] and [+16 .. +23] -> two 16B LDS loads.
__device__ __forceinline__ v16bf load_a_frag(const u16* lA, int mrow, int ldk, int k0, int lane) {
  int base = mrow * ldk + k0 + ((lane >> 4) << 3);
  ABits t;
  t.q[0] = *(const uint4*)(lA + base);
  t.q[1] = *(const uint4*)(lA + base + 16);
  return t.v;
}

// B fragment (32x16 bf16, cols striped over lanes): lane L holds column
// n = L%16, K-chunks [k0 + (L>=16?8:0) .. +7] and [+16 .. +23].
// Weights are [E][N][K] row-major -> two 16B loads per lane, 16 elements apart.
__device__ __forceinline__ v16bf load_b_frag(const u16* __restrict__ wb, long nrow, int K, int k0, int lane) {
  const u16* p = wb + nrow * (long)K + k0 + ((lane >> 4) << 3);
  ABits t;
  t.q[0] = *(const uint4*)(p);
  t.q[1] = *(const uint4*)(p + 16);
  return t.v;
}

__device__ __forceinline__ v8f wmma_bf16(v16bf a, v16bf b, v8f c) {
  return __builtin_amdgcn_wmma_f32_16x16x32_bf16(false, a, false, b, (short)0, c, false, false);
}

// One gate-blended layer: acc[mi][ni] = sum_e gate[m,e] * (A @ W_e^T) tile.
// Wave tile = 32 rows (mi=0,1) x (NI*16) cols starting at nBase.
template<int KDIM, int NI>
__device__ __forceinline__ void gemm_blend(const u16* lA, int ldk,
                                           const u16* __restrict__ wb, int N,
                                           const float* lG,
                                           int lane, int nBase,
                                           v8f acc[2][NI]) {
  const int lo  = lane & 15;
  const int hi8 = (lane >> 4) << 3;
#pragma unroll
  for (int mi = 0; mi < 2; ++mi)
#pragma unroll
    for (int ni = 0; ni < NI; ++ni)
#pragma unroll
      for (int r = 0; r < 8; ++r) acc[mi][ni][r] = 0.f;

  for (int e = 0; e < 3; ++e) {
    v8f p[2][NI];
#pragma unroll
    for (int mi = 0; mi < 2; ++mi)
#pragma unroll
      for (int ni = 0; ni < NI; ++ni)
#pragma unroll
        for (int r = 0; r < 8; ++r) p[mi][ni][r] = 0.f;

    for (int k0 = 0; k0 < KDIM; k0 += 32) {
      v16bf a0 = load_a_frag(lA, lo,      ldk, k0, lane);
      v16bf a1 = load_a_frag(lA, 16 + lo, ldk, k0, lane);
#pragma unroll
      for (int ni = 0; ni < NI; ++ni) {
        long nrow = (long)e * N + nBase + ni * 16 + lo;
        v16bf b = load_b_frag(wb, nrow, KDIM, k0, lane);
        p[0][ni] = wmma_bf16(a0, b, p[0][ni]);
        p[1][ni] = wmma_bf16(a1, b, p[1][ni]);
      }
    }
    // blend with per-row gate: C layout -> element r of a tile is row
    // m = mi*16 + r + (lane>=16 ? 8 : 0), col n = base + lane%16.
#pragma unroll
    for (int mi = 0; mi < 2; ++mi)
#pragma unroll
      for (int r = 0; r < 8; ++r) {
        int m = mi * 16 + r + hi8;
        float s = lG[m * 3 + e];
#pragma unroll
        for (int ni = 0; ni < NI; ++ni) acc[mi][ni][r] += s * p[mi][ni][r];
      }
  }
}

// bias-blend + relu + bf16 restage into LDS (stride LDA2) for next layer
__device__ __forceinline__ void epilogue_relu_store(u16* lA, const float* lG,
                                                    const float* __restrict__ bias,
                                                    v8f acc[2][4], int lane, int nBase) {
  const int lo  = lane & 15;
  const int hi8 = (lane >> 4) << 3;
  float bb[4][3];
#pragma unroll
  for (int ni = 0; ni < 4; ++ni)
#pragma unroll
    for (int e = 0; e < 3; ++e)
      bb[ni][e] = bias[e * 512 + nBase + ni * 16 + lo];

  __syncthreads();   // all waves finished reading current-layer activations
#pragma unroll
  for (int mi = 0; mi < 2; ++mi)
#pragma unroll
    for (int r = 0; r < 8; ++r) {
      int m = mi * 16 + r + hi8;
      float g0 = lG[m * 3 + 0], g1 = lG[m * 3 + 1], g2 = lG[m * 3 + 2];
#pragma unroll
      for (int ni = 0; ni < 4; ++ni) {
        float v = acc[mi][ni][r] + g0 * bb[ni][0] + g1 * bb[ni][1] + g2 * bb[ni][2];
        v = fmaxf(v, 0.f);
        lA[m * LDA2 + nBase + ni * 16 + lo] = f2bf_rne(v);
      }
    }
  __syncthreads();   // next layer may read
}

__global__ __launch_bounds__(256) void moe_fused_kernel(
    const float* __restrict__ f1, const float* __restrict__ f2, const float* __restrict__ lat,
    const u16* __restrict__ wb1, const u16* __restrict__ wb2,
    const u16* __restrict__ wb3, const u16* __restrict__ wb4,
    const float* __restrict__ b1, const float* __restrict__ b2,
    const float* __restrict__ b3, const float* __restrict__ b4,
    const float* __restrict__ gate, float* __restrict__ out) {
  __shared__ u16  lA[MT * LDA1];
  __shared__ float lG[MT * 3];
  const int tid  = threadIdx.x;
  const int lane = tid & 31;
  const int wave = tid >> 5;
  const long r0  = (long)blockIdx.x * MT;

  // stage gates + x (fp32 -> bf16) into LDS
  for (int i = tid; i < MT * 3; i += 256) lG[i] = gate[r0 * 3 + i];
  for (int i = tid; i < MT * 640; i += 256) {
    int m = i / 640, k = i - m * 640;
    float v;
    if (k < 256)      v = f1[(r0 + m) * 256 + k];
    else if (k < 512) v = f2[(r0 + m) * 256 + (k - 256)];
    else              v = lat[(r0 + m) * 128 + (k - 512)];
    lA[m * LDA1 + k] = f2bf_rne(v);
  }
  __syncthreads();

  const int lo    = lane & 15;
  const int hi8   = (lane >> 4) << 3;
  const int nBase = wave * 64;   // 8 waves x 64 cols = 512

  { // layer 1: K=640 -> N=512
    v8f acc[2][4];
    gemm_blend<640, 4>(lA, LDA1, wb1, 512, lG, lane, nBase, acc);
    epilogue_relu_store(lA, lG, b1, acc, lane, nBase);
  }
  { // layer 2: 512 -> 512
    v8f acc[2][4];
    gemm_blend<512, 4>(lA, LDA2, wb2, 512, lG, lane, nBase, acc);
    epilogue_relu_store(lA, lG, b2, acc, lane, nBase);
  }
  { // layer 3: 512 -> 512
    v8f acc[2][4];
    gemm_blend<512, 4>(lA, LDA2, wb3, 512, lG, lane, nBase, acc);
    epilogue_relu_store(lA, lG, b3, acc, lane, nBase);
  }
  { // layer 4: 512 -> 128, fp32 out, no relu
    v8f acc[2][1];
    const int nB4 = wave * 16;   // 8 waves x 16 cols = 128
    gemm_blend<512, 1>(lA, LDA2, wb4, 128, lG, lane, nB4, acc);
    float bb[3];
#pragma unroll
    for (int e = 0; e < 3; ++e) bb[e] = b4[e * 128 + nB4 + lo];
#pragma unroll
    for (int mi = 0; mi < 2; ++mi)
#pragma unroll
      for (int r = 0; r < 8; ++r) {
        int m = mi * 16 + r + hi8;
        float g0 = lG[m * 3 + 0], g1 = lG[m * 3 + 1], g2 = lG[m * 3 + 2];
        float v = acc[mi][0][r] + g0 * bb[0] + g1 * bb[1] + g2 * bb[2];
        out[(r0 + m) * 128 + nB4 + lo] = v;
      }
  }
}

// ---------------------------------------------------------------------------
// gating MLP: softmax(g2(elu(g1(elu(g0(x)))))), 16 rows per wave, one at a time
// ---------------------------------------------------------------------------
__device__ __forceinline__ float eluf(float v) { return v > 0.f ? v : (__expf(v) - 1.f); }

__global__ __launch_bounds__(256) void gate_kernel(
    const float* __restrict__ f1, const float* __restrict__ f2, const float* __restrict__ lat,
    const float* __restrict__ g0w, const float* __restrict__ g0b,
    const float* __restrict__ g1w, const float* __restrict__ g1b,
    const float* __restrict__ g2w, const float* __restrict__ g2b,
    float* __restrict__ gate) {
  __shared__ float sg1[32 * 32];
  __shared__ float sg2[3 * 32];
  __shared__ float sb0[32], sb1[32], sb2[3];
  const int tid = threadIdx.x, lane = tid & 31, wave = tid >> 5;
  for (int i = tid; i < 1024; i += 256) sg1[i] = g1w[i];
  for (int i = tid; i < 96;   i += 256) sg2[i] = g2w[i];
  if (tid < 32) { sb0[tid] = g0b[tid]; sb1[tid] = g1b[tid]; }
  if (tid < 3)  sb2[tid] = g2b[tid];
  __syncthreads();

  const int ROWS = 16;
  long rowBase = ((long)blockIdx.x * 8 + wave) * ROWS;
  const float* w = g0w + lane * 640;
  for (int i = 0; i < ROWS; ++i) {
    long row = rowBase + i;
    const float* x1 = f1 + row * 256;
    const float* x2 = f2 + row * 256;
    const float* xl = lat + row * 128;
    float acc = sb0[lane];
    for (int k = 0; k < 256; k += 4) {
      float4 a = *(const float4*)(x1 + k); float4 b = *(const float4*)(w + k);
      acc += a.x * b.x + a.y * b.y + a.z * b.z + a.w * b.w;
    }
    for (int k = 0; k < 256; k += 4) {
      float4 a = *(const float4*)(x2 + k); float4 b = *(const float4*)(w + 256 + k);
      acc += a.x * b.x + a.y * b.y + a.z * b.z + a.w * b.w;
    }
    for (int k = 0; k < 128; k += 4) {
      float4 a = *(const float4*)(xl + k); float4 b = *(const float4*)(w + 512 + k);
      acc += a.x * b.x + a.y * b.y + a.z * b.z + a.w * b.w;
    }
    float h0 = eluf(acc);
    float a1 = sb1[lane];
    for (int k = 0; k < 32; ++k) a1 += __shfl(h0, k, 32) * sg1[lane * 32 + k];
    float h1 = eluf(a1);
    float l0 = sb2[0], l1 = sb2[1], l2 = sb2[2];
    for (int j = 0; j < 32; ++j) {
      float hj = __shfl(h1, j, 32);
      l0 += hj * sg2[j]; l1 += hj * sg2[32 + j]; l2 += hj * sg2[64 + j];
    }
    float mx = fmaxf(l0, fmaxf(l1, l2));
    float e0 = __expf(l0 - mx), e1 = __expf(l1 - mx), e2 = __expf(l2 - mx);
    float inv = 1.f / (e0 + e1 + e2);
    if (lane == 0) {
      gate[row * 3 + 0] = e0 * inv;
      gate[row * 3 + 1] = e1 * inv;
      gate[row * 3 + 2] = e2 * inv;
    }
  }
}

// ---------------------------------------------------------------------------
// fp32 -> bf16 weight conversion into workspace (layout preserved [E][N][K])
// ---------------------------------------------------------------------------
__global__ void cvt_bf16(const float* __restrict__ src, u16* __restrict__ dst, int n) {
  int i = blockIdx.x * blockDim.x + threadIdx.x;
  int stride = gridDim.x * blockDim.x;
  for (; i < n; i += stride) dst[i] = f2bf_rne(src[i]);
}

// ---------------------------------------------------------------------------
extern "C" void kernel_launch(void* const* d_in, const int* in_sizes, int n_in,
                              void* d_out, int out_size, void* d_ws, size_t ws_size,
                              hipStream_t stream) {
  const float* f1  = (const float*)d_in[0];
  const float* f2  = (const float*)d_in[1];
  const float* lat = (const float*)d_in[2];
  const float* g0w = (const float*)d_in[3];
  const float* g0b = (const float*)d_in[4];
  const float* g1w = (const float*)d_in[5];
  const float* g1b = (const float*)d_in[6];
  const float* g2w = (const float*)d_in[7];
  const float* g2b = (const float*)d_in[8];
  const float* W1  = (const float*)d_in[9];
  const float* b1  = (const float*)d_in[10];
  const float* W2  = (const float*)d_in[11];
  const float* b2  = (const float*)d_in[12];
  const float* W3  = (const float*)d_in[13];
  const float* b3  = (const float*)d_in[14];
  const float* W4  = (const float*)d_in[15];
  const float* b4  = (const float*)d_in[16];

  float* out  = (float*)d_out;
  float* gate = out + (size_t)B_TOT * 128;   // second output slot, also consumed by fused kernel

  const int nW1 = 3 * 512 * 640;   // 983040
  const int nW2 = 3 * 512 * 512;   // 786432
  const int nW3 = 3 * 512 * 512;   // 786432
  const int nW4 = 3 * 128 * 512;   // 196608
  u16* wb1 = (u16*)d_ws;
  u16* wb2 = wb1 + nW1;
  u16* wb3 = wb2 + nW2;
  u16* wb4 = wb3 + nW3;

  cvt_bf16<<<1024, 256, 0, stream>>>(W1, wb1, nW1);
  cvt_bf16<<<1024, 256, 0, stream>>>(W2, wb2, nW2);
  cvt_bf16<<<1024, 256, 0, stream>>>(W3, wb3, nW3);
  cvt_bf16<<<512,  256, 0, stream>>>(W4, wb4, nW4);

  gate_kernel<<<B_TOT / (8 * 16), 256, 0, stream>>>(f1, f2, lat, g0w, g0b, g1w, g1b, g2w, g2b, gate);

  moe_fused_kernel<<<B_TOT / MT, 256, 0, stream>>>(f1, f2, lat,
                                                   wb1, wb2, wb3, wb4,
                                                   b1, b2, b3, b4,
                                                   gate, out);
}
